// CouplingLayerBlock_RQS_6158983102915
// MI455X (gfx1250) — matrix operand order repaired and checked by reading, not compiled, not run on hardware
//
#include <hip/hip_runtime.h>

typedef __attribute__((ext_vector_type(16))) _Float16 v16h;
typedef __attribute__((ext_vector_type(8)))  _Float16 v8h;
typedef __attribute__((ext_vector_type(8)))  float    v8f;

#define MTILE 128
#define NTH   256
#define HP    136   // padded f16 row stride (128 data + 8 pad) -> conflict-free frag loads
#define PSP   50    // padded f32 row stride for params (48 used)
#define TAILB 12.0f
#define MIN_BWH 1e-6f
#define MIN_DD  1e-6f

__device__ __forceinline__ void softmax_edges(const float u[8], float cw[9]) {
  float m = u[0];
#pragma unroll
  for (int k = 1; k < 8; ++k) m = fmaxf(m, u[k]);
  float e[8]; float s = 0.f;
#pragma unroll
  for (int k = 0; k < 8; ++k) { e[k] = __expf(u[k] - m); s += e[k]; }
  float inv = (1.f - 8.f * MIN_BWH) / s;
  float c = 0.f;
  cw[0] = -TAILB;
#pragma unroll
  for (int k = 0; k < 8; ++k) {
    c += MIN_BWH + e[k] * inv;
    cw[k + 1] = 2.f * TAILB * c - TAILB;
  }
  cw[8] = TAILB;
}

__global__ __launch_bounds__(NTH) void flow_rqs_kernel(
    const float* __restrict__ gin, const float* __restrict__ gcond,
    const float* __restrict__ gW1, const float* __restrict__ gb1,
    const float* __restrict__ gW2, const float* __restrict__ gb2,
    const float* __restrict__ gW3, const float* __restrict__ gb3,
    float* __restrict__ gout_x, float* __restrict__ gout_ld)
{
  __shared__ float xs[MTILE][4];
  __shared__ float cs[MTILE];
  __shared__ float ldsum[MTILE];
  __shared__ float ladbuf[MTILE][2];
  __shared__ __align__(16) _Float16 h1[MTILE * HP];
  __shared__ __align__(16) _Float16 h2[MTILE * HP];
  __shared__ __align__(16) _Float16 w2s[128 * HP];
  __shared__ __align__(16) _Float16 w3s[48 * HP];
  __shared__ float w1s[128][3];
  __shared__ float b1s[128], b2s[128], b3s[48];
  __shared__ float ps[MTILE * PSP];

  const int tid  = threadIdx.x;
  const int wave = tid >> 5, lane = tid & 31;
  const int nl   = lane & 15, half = lane >> 4;
  const long row0 = (long)blockIdx.x * MTILE;

  const int midx0[4] = {0, 1, 0, 2};
  const int midx1[4] = {2, 3, 1, 3};
  const int iidx0[4] = {1, 0, 2, 0};
  const int iidx1[4] = {3, 2, 3, 1};

  // ---- load batch tile ----
#pragma unroll
  for (int i = tid; i < MTILE * 4; i += NTH) xs[i >> 2][i & 3] = gin[row0 * 4 + i];
  if (tid < MTILE) { cs[tid] = gcond[row0 + tid]; ldsum[tid] = 0.f; }
  __syncthreads();

  for (int l = 0; l < 4; ++l) {
    // ---- stage weights to LDS (W2/W3 as f16 for WMMA) ----
    const float* W2l = gW2 + l * 128 * 128;
    for (int i = tid; i < 128 * 128; i += NTH)
      w2s[(i >> 7) * HP + (i & 127)] = (_Float16)W2l[i];
    const float* W3l = gW3 + l * 46 * 128;
    for (int i = tid; i < 48 * 128; i += NTH) {
      int n = i >> 7;
      w3s[n * HP + (i & 127)] = (n < 46) ? (_Float16)W3l[i] : (_Float16)0.f;
    }
    for (int i = tid; i < 128 * 3; i += NTH) ((float*)w1s)[i] = gW1[l * 384 + i];
    if (tid < 128) { b1s[tid] = gb1[l * 128 + tid]; b2s[tid] = gb2[l * 128 + tid]; }
    if (tid < 48)  b3s[tid] = (tid < 46) ? gb3[l * 46 + tid] : 0.f;
    __syncthreads();

    // ---- stage 1: h1 = relu([x_m0, x_m1, cond] @ W1^T + b1)  (K=3, VALU) ----
    {
      int r = tid >> 1;
      float a0 = xs[r][midx0[l]], a1 = xs[r][midx1[l]], a2 = cs[r];
      int c0 = (tid & 1) * 64;
#pragma unroll
      for (int j = 0; j < 64; ++j) {
        int c = c0 + j;
        float v = fmaf(a0, w1s[c][0], fmaf(a1, w1s[c][1], fmaf(a2, w1s[c][2], b1s[c])));
        h1[r * HP + c] = (_Float16)fmaxf(v, 0.f);
      }
    }
    __syncthreads();

    // ---- stage 2: h2 = relu(h1 @ W2^T + b2), WMMA f16 -> f32 acc ----
    {
      const int n0 = wave * 16;
      v16h bfrag[4];
#pragma unroll
      for (int kb = 0; kb < 4; ++kb) {
        const _Float16* bp = &w2s[(n0 + nl) * HP + kb * 32 + half * 16];
        v8h blo = *(const v8h*)bp;
        v8h bhi = *(const v8h*)(bp + 8);
        bfrag[kb] = __builtin_shufflevector(blo, bhi, 0,1,2,3,4,5,6,7,8,9,10,11,12,13,14,15);
      }
      float bn = b2s[n0 + nl];
#pragma unroll
      for (int mt = 0; mt < 8; ++mt) {
        const int m0 = mt * 16;
        v8f acc = {};
#pragma unroll
        for (int kb = 0; kb < 4; ++kb) {
          const _Float16* ap = &h1[(m0 + nl) * HP + kb * 32 + half * 16];
          v8h alo = *(const v8h*)ap;
          v8h ahi = *(const v8h*)(ap + 8);
          v16h afrag = __builtin_shufflevector(alo, ahi, 0,1,2,3,4,5,6,7,8,9,10,11,12,13,14,15);
          acc = __builtin_amdgcn_wmma_f32_16x16x32_f16(false, afrag, false, bfrag[kb],
                                                       (short)0, acc, false, false);
        }
#pragma unroll
        for (int rI = 0; rI < 8; ++rI) {
          float v = fmaxf(acc[rI] + bn, 0.f);
          h2[(m0 + half * 8 + rI) * HP + n0 + nl] = (_Float16)v;
        }
      }
    }
    __syncthreads();

    // ---- stage 3: params = h2 @ W3^T + b3  (N padded 46 -> 48) ----
#pragma unroll
    for (int i = 0; i < 3; ++i) {
      int t  = wave + 8 * i;
      int m0 = (t & 7) * 16, n0 = (t >> 3) * 16;
      v8f acc = {};
#pragma unroll
      for (int kb = 0; kb < 4; ++kb) {
        const _Float16* ap = &h2[(m0 + nl) * HP + kb * 32 + half * 16];
        v8h alo = *(const v8h*)ap;
        v8h ahi = *(const v8h*)(ap + 8);
        v16h afrag = __builtin_shufflevector(alo, ahi, 0,1,2,3,4,5,6,7,8,9,10,11,12,13,14,15);
        const _Float16* bp = &w3s[(n0 + nl) * HP + kb * 32 + half * 16];
        v8h blo = *(const v8h*)bp;
        v8h bhi = *(const v8h*)(bp + 8);
        v16h bfrag = __builtin_shufflevector(blo, bhi, 0,1,2,3,4,5,6,7,8,9,10,11,12,13,14,15);
        acc = __builtin_amdgcn_wmma_f32_16x16x32_f16(false, afrag, false, bfrag,
                                                     (short)0, acc, false, false);
      }
      float bn = b3s[n0 + nl];
#pragma unroll
      for (int rI = 0; rI < 8; ++rI)
        ps[(m0 + half * 8 + rI) * PSP + n0 + nl] = acc[rI] + bn;
    }
    __syncthreads();

    // ---- stage 4: RQS spline, one thread per (row, transformed dim) ----
    {
      int row = tid >> 1, f = tid & 1;
      const float* p = &ps[row * PSP];
      int xi = f ? iidx1[l] : iidx0[l];
      float x = xs[row][xi];
      bool inside = (x >= -TAILB) && (x <= TAILB);
      float xc = fminf(fmaxf(x, -TAILB), TAILB);

      float uw[8], uh[8];
#pragma unroll
      for (int k = 0; k < 8; ++k) { uw[k] = p[2 * k + f]; uh[k] = p[16 + 2 * k + f]; }
      float cw[9], ch[9];
      softmax_edges(uw, cw);
      softmax_edges(uh, ch);
      float d[9];
      d[0] = 1.f; d[8] = 1.f;   // pad const -> softplus gives exactly 1 - MIN_D, + MIN_D
#pragma unroll
      for (int k = 0; k < 7; ++k) {
        float u = p[32 + 2 * k + f];
        d[k + 1] = MIN_DD + (fmaxf(u, 0.f) + log1pf(__expf(-fabsf(u))));
      }
      // branchless bin search + gather (edges are strictly increasing)
      float icw = cw[0], icw1 = cw[1], ich = ch[0], ich1 = ch[1], dk = d[0], dk1 = d[1];
#pragma unroll
      for (int k = 1; k < 8; ++k) {
        if (xc >= cw[k]) {
          icw = cw[k]; icw1 = cw[k + 1];
          ich = ch[k]; ich1 = ch[k + 1];
          dk  = d[k];  dk1  = d[k + 1];
        }
      }
      float iw = icw1 - icw, ih = ich1 - ich;
      float delta = ih / iw;
      float theta = (xc - icw) / iw;
      float omt = 1.f - theta;
      float tomt = theta * omt;
      float num = ih * (delta * theta * theta + dk * tomt);
      float den = delta + (dk + dk1 - 2.f * delta) * tomt;
      float y = ich + num / den;
      float dnum = delta * delta * (dk1 * theta * theta + 2.f * delta * tomt + dk * omt * omt);
      float lad = __logf(dnum) - 2.f * __logf(den);
      xs[row][xi] = inside ? y : x;
      ladbuf[row][f] = inside ? lad : 0.f;
    }
    __syncthreads();
    if (tid < MTILE) ldsum[tid] += ladbuf[tid][0] + ladbuf[tid][1];
    __syncthreads();
  }

  // ---- write outputs: x [B,4] then logdets [B,1] ----
#pragma unroll
  for (int i = tid; i < MTILE * 4; i += NTH) gout_x[row0 * 4 + i] = xs[i >> 2][i & 3];
  if (tid < MTILE) gout_ld[row0 + tid] = ldsum[tid];
}

extern "C" void kernel_launch(void* const* d_in, const int* in_sizes, int n_in,
                              void* d_out, int out_size, void* d_ws, size_t ws_size,
                              hipStream_t stream) {
  const float* inputs = (const float*)d_in[0];
  const float* cond   = (const float*)d_in[1];
  const float* W1     = (const float*)d_in[2];
  const float* b1     = (const float*)d_in[3];
  const float* W2     = (const float*)d_in[4];
  const float* b2     = (const float*)d_in[5];
  const float* W3     = (const float*)d_in[6];
  const float* b3     = (const float*)d_in[7];
  int B = in_sizes[0] / 4;
  float* out_x  = (float*)d_out;
  float* out_ld = (float*)d_out + (size_t)B * 4;
  int grid = B / MTILE;
  flow_rqs_kernel<<<grid, NTH, 0, stream>>>(inputs, cond, W1, b1, W2, b2, W3, b3,
                                            out_x, out_ld);
}
